// NetFFM_31679678775363
// MI455X (gfx1250) — compile-verified, matrix operand order checked by reading, not run on hardware
//
#include <hip/hip_runtime.h>
#include <hip/hip_bf16.h>

#define NUM_INPUTS 24
#define FACTOR     16
#define BATCH      8192
#define PAIRS      276
#define HIDDEN     64

typedef __bf16 bfx4  __attribute__((ext_vector_type(4)));
typedef __bf16 bfx8  __attribute__((ext_vector_type(8)));
typedef __bf16 v16bf __attribute__((ext_vector_type(16)));
typedef float  v8f   __attribute__((ext_vector_type(8)));

// ---------------------------------------------------------------------------
// Pre-kernel: W1 [P, 2F=32, H=64] f32  ->  W1T [P, H=64, 2F=32] bf16 in d_ws.
// Gives each WMMA B-fragment lane two contiguous 16B chunks (K-contiguous).
// ---------------------------------------------------------------------------
__global__ void convert_w1_kernel(const float* __restrict__ W1,
                                  __bf16* __restrict__ W1T) {
  int t = blockIdx.x * blockDim.x + threadIdx.x;
  if (t >= PAIRS * HIDDEN * 2 * FACTOR) return;
  int p   = t >> 11;        // / (64*32)
  int rem = t & 2047;
  int n   = rem >> 5;       // hidden index
  int k   = rem & 31;       // concat-input index
  W1T[t] = (__bf16)W1[p * (2 * FACTOR * HIDDEN) + k * HIDDEN + n];
}

// Padded LDS row stride (bf16 units): 24*16 + 8 -> spreads banks across lanes
#define XS_STRIDE (NUM_INPUTS * FACTOR + 8)   // 392

// ---------------------------------------------------------------------------
// Main kernel: one block = one 16-row batch tile; 8 waves split the 276 pairs.
// Per pair: [16x32]bf16 @ [32x64]bf16 via 4x v_wmma_f32_16x16x32_bf16 with
// inline-zero C, then bias+relu+W2-dot epilogue accumulated per row.
// Pair -> (i, j) decode comes from a per-block LDS table (no divergent carry
// loop in the hot path).
// ---------------------------------------------------------------------------
__global__ __launch_bounds__(256) void ffm_wmma_kernel(
    const int*   __restrict__ inputs,  // [B, 24] int32
    const float* __restrict__ w,       // [V, 1]
    const float* __restrict__ v,       // [V, 16]
    const float* __restrict__ b,       // [1]
    const __bf16* __restrict__ W1T,    // [P, 64, 32] bf16 (pre-transposed)
    const float* __restrict__ b1,      // [P, 64]
    const float* __restrict__ W2,      // [P, 64]
    const float* __restrict__ b2,      // [P, 1]
    float* __restrict__ out) {         // [B]
  __shared__ __align__(16) __bf16 xs[16 * XS_STRIDE];  // factor tile (bf16)
  __shared__ int            sidx[16 * NUM_INPUTS];
  __shared__ unsigned short spair[PAIRS];              // (i << 5) | j
  __shared__ float lsum[16];
  __shared__ float partial[8][16];
  __shared__ float bpart[8];

  const int tid  = threadIdx.x;
  const int base = blockIdx.x * 16;    // first batch row of this tile

  if (tid < 16) lsum[tid] = 0.f;

  // ---- build pair -> (i,j) decode table (setup phase only) ----
  for (int e = tid; e < PAIRS; e += 256) {
    int i = 0, r = e;
    while (r >= NUM_INPUTS - 1 - i) {
      r -= NUM_INPUTS - 1 - i;
      ++i;
    }
    spair[e] = (unsigned short)((i << 5) | (i + 1 + r));
  }
  __syncthreads();

  // ---- stage indices + linear term (sum of w over fields, per row) ----
  for (int e = tid; e < 16 * NUM_INPUTS; e += 256) {
    int idx = inputs[base * NUM_INPUTS + e];   // e = row*24 + field
    sidx[e] = idx;
    atomicAdd(&lsum[e / NUM_INPUTS], w[idx]);  // ds_add_f32
  }
  __syncthreads();

  // ---- gather factor embeddings, convert f32 -> bf16 into LDS tile ----
  const float4* v4 = (const float4*)v;
  for (int e = tid; e < 16 * NUM_INPUTS * 4; e += 256) {
    int q = e >> 2, c = e & 3;
    int row = q / NUM_INPUTS, field = q % NUM_INPUTS;
    float4 val = v4[(size_t)sidx[q] * 4 + c];
    bfx4 hv = { (__bf16)val.x, (__bf16)val.y, (__bf16)val.z, (__bf16)val.w };
    *(bfx4*)&xs[row * XS_STRIDE + field * FACTOR + c * 4] = hv;
  }
  __syncthreads();

  // ---- pair loop: wave `wv` handles p = wv, wv+8, wv+16, ... ----
  const int wave  = tid >> 5;
  const int lane  = tid & 31;
  const int nlane = lane & 15;
  const int khalf = (lane >> 4) << 3;  // 0 for lanes 0-15, 8 for lanes 16-31
  const int arow  = nlane * XS_STRIDE;

  float rowacc[8] = {0.f, 0.f, 0.f, 0.f, 0.f, 0.f, 0.f, 0.f};
  float bacc = 0.f;
  const v8f zeroc = {0.f, 0.f, 0.f, 0.f, 0.f, 0.f, 0.f, 0.f};

  for (int p = wave; p < PAIRS; p += 8) {
    const int ij = spair[p];             // one ds_load, no divergent decode
    const int fi = ij >> 5;
    const int fj = ij & 31;

    // A fragment: rows = batch rows (M), K = [xv_i | xv_j]
    // 16-bit A layout: lanes<16 hold K {0..7,16..23}, lanes>=16 hold
    // K {8..15,24..31} -> two contiguous bf16x8 chunks per field.
    bfx8 a0 = *(const bfx8*)&xs[arow + fi * FACTOR + khalf];
    bfx8 a1 = *(const bfx8*)&xs[arow + fj * FACTOR + khalf];
    v16bf a = __builtin_shufflevector(a0, a1, 0, 1, 2, 3, 4, 5, 6, 7,
                                      8, 9, 10, 11, 12, 13, 14, 15);

    const __bf16* wp = W1T + (size_t)p * (HIDDEN * 2 * FACTOR);
    v8f acc[4];
#pragma unroll
    for (int nt = 0; nt < 4; ++nt) {
      int n = nt * 16 + nlane;
      // B fragment: column n, same K halves as A (mirrored layout)
      bfx8 bA = *(const bfx8*)&wp[n * 32 + khalf];
      bfx8 bB = *(const bfx8*)&wp[n * 32 + 16 + khalf];
      v16bf bb = __builtin_shufflevector(bA, bB, 0, 1, 2, 3, 4, 5, 6, 7,
                                         8, 9, 10, 11, 12, 13, 14, 15);
      // 8 args: (neg_a, A, neg_b, B, c_mod, C, reuse_a, reuse_b)
      // C = 0 -> inline-constant SRC2, no splat movs, no load dependency
      acc[nt] = __builtin_amdgcn_wmma_f32_16x16x32_bf16(
          false, a, false, bb, (short)0, zeroc, false, false);
    }

    // Epilogue: +b1, relu, dot with W2 column; C layout:
    // vgpr r, lane l -> M = r + 8*(l>=16), N = nt*16 + (l&15)
    const int pbase = p * HIDDEN;
#pragma unroll
    for (int nt = 0; nt < 4; ++nt) {
      int n = nt * 16 + nlane;
      float bv  = b1[pbase + n];
      float w2v = W2[pbase + n];
#pragma unroll
      for (int r = 0; r < 8; ++r) {
        float h = fmaxf(acc[nt][r] + bv, 0.f);   // single v_max_num_f32
        rowacc[r] = fmaf(h, w2v, rowacc[r]);
      }
    }
    bacc += b2[p];
  }

  // ---- reduce over N (16 lanes per half) ----
#pragma unroll
  for (int r = 0; r < 8; ++r)
    for (int mask = 1; mask < 16; mask <<= 1)
      rowacc[r] += __shfl_xor(rowacc[r], mask, 32);

  if (lane == 0) {
#pragma unroll
    for (int r = 0; r < 8; ++r) partial[wave][r] = rowacc[r];      // rows 0..7
    bpart[wave] = bacc;
  }
  if (lane == 16) {
#pragma unroll
    for (int r = 0; r < 8; ++r) partial[wave][8 + r] = rowacc[r];  // rows 8..15
  }
  __syncthreads();

  // ---- combine waves, add linear + biases, sigmoid ----
  if (tid < 16) {
    float s = lsum[tid] + b[0];
    float bs = 0.f;
#pragma unroll
    for (int wv = 0; wv < 8; ++wv) {
      s  += partial[wv][tid];
      bs += bpart[wv];
    }
    s += bs;  // sum of b2 over pairs (each pair counted once across waves)
    out[base + tid] = 1.f / (1.f + __expf(-s));
  }
}

extern "C" void kernel_launch(void* const* d_in, const int* in_sizes, int n_in,
                              void* d_out, int out_size, void* d_ws,
                              size_t ws_size, hipStream_t stream) {
  (void)in_sizes; (void)n_in; (void)out_size; (void)ws_size;
  const int*   inputs = (const int*)d_in[0];
  const float* w      = (const float*)d_in[1];
  const float* v      = (const float*)d_in[2];
  const float* b      = (const float*)d_in[3];
  const float* W1     = (const float*)d_in[4];
  const float* b1     = (const float*)d_in[5];
  const float* W2     = (const float*)d_in[6];
  const float* b2     = (const float*)d_in[7];
  float* out = (float*)d_out;
  __bf16* W1T = (__bf16*)d_ws;  // 276*64*32*2 B ~= 1.1 MB scratch

  int nconv = PAIRS * HIDDEN * 2 * FACTOR;
  convert_w1_kernel<<<(nconv + 255) / 256, 256, 0, stream>>>(W1, W1T);
  ffm_wmma_kernel<<<BATCH / 16, 256, 0, stream>>>(inputs, w, v, b, W1T, b1,
                                                  W2, b2, out);
}
